// OrthoLinear_8194797601329
// MI455X (gfx1250) — matrix operand order repaired and checked by reading, not compiled, not run on hardware
//
#include <hip/hip_runtime.h>
#include <cstdint>

typedef __attribute__((ext_vector_type(16))) _Float16 v16h;
typedef __attribute__((ext_vector_type(8)))  float    v8f;
typedef __attribute__((ext_vector_type(4)))  int      v4i;

#define IN_DIM   4096
#define OUT_DIM  11008
#define TOKENS   2048
#define HALF_IN  (IN_DIM / 2)

// ---------------------------------------------------------------------------
// Kernel 1: cast x (fp32) -> xh (f16), 4 elements per thread
// ---------------------------------------------------------------------------
__global__ __launch_bounds__(256) void xcast_kernel(const float* __restrict__ x,
                                                    _Float16* __restrict__ xh) {
    const int i = blockIdx.x * blockDim.x + threadIdx.x;   // 4 floats per thread
    const float4 f = ((const float4*)x)[i];
    union { _Float16 h[4]; uint2 u; } cvt;
    cvt.h[0] = (_Float16)f.x;
    cvt.h[1] = (_Float16)f.y;
    cvt.h[2] = (_Float16)f.z;
    cvt.h[3] = (_Float16)f.w;
    ((uint2*)xh)[i] = cvt.u;
}

// ---------------------------------------------------------------------------
// Kernel 2: dequant 4-bit packed weights -> f16 W [OUT_DIM, IN_DIM]
// each thread: 4 packed int32 (NT load, read-once stream) -> 8 f16
// ---------------------------------------------------------------------------
__global__ __launch_bounds__(256) void dequant_kernel(const int* __restrict__ packed,
                                                      const float* __restrict__ scales,
                                                      _Float16* __restrict__ W) {
    const int tid = blockIdx.x * blockDim.x + threadIdx.x;      // OUT*HALF/4 threads
    const int r  = tid / (HALF_IN / 4);
    const int j4 = (tid % (HALF_IN / 4)) * 4;                   // packed column
    const v4i p = __builtin_nontemporal_load(
        (const v4i*)(packed + (size_t)r * HALF_IN + j4));
    const float s = scales[r];
    union { _Float16 h[8]; uint4 u; } out;
#pragma unroll
    for (int q = 0; q < 4; ++q) {
        const int v = p[q];
        out.h[2 * q]     = (_Float16)((float)((v & 0xF) - 8) * s);
        out.h[2 * q + 1] = (_Float16)((float)(((v >> 4) & 0xF) - 8) * s);
    }
    *(uint4*)(W + (size_t)r * IN_DIM + 2 * j4) = out.u;   // keep W hot in L2
}

// ---------------------------------------------------------------------------
// Kernel 3: deterministic sparse COO add into W (rows sorted; duplicates
// summed in index order by the first occurrence; plain b16 store, no atomics)
// ---------------------------------------------------------------------------
__global__ __launch_bounds__(256) void scatter_kernel(const float* __restrict__ vals,
                                                      const int* __restrict__ rows,
                                                      const int* __restrict__ cols,
                                                      _Float16* __restrict__ W, int nnz) {
    const int i = blockIdx.x * blockDim.x + threadIdx.x;
    if (i >= nnz) return;
    const int r = rows[i];
    const int c = cols[i];
    // only the first occurrence of (r,c) within the (contiguous) row segment writes
    for (int j = i - 1; j >= 0 && rows[j] == r; --j)
        if (cols[j] == c) return;
    float sum = vals[i];
    for (int j = i + 1; j < nnz && rows[j] == r; ++j)
        if (cols[j] == c) sum += vals[j];
    const size_t idx = (size_t)r * IN_DIM + c;
    W[idx] = (_Float16)((float)W[idx] + sum);   // ALPHA == 1.0
}

// ---------------------------------------------------------------------------
// Kernel 4: WMMA GEMM  C[M,N] = Xh[M,K] * W[N,K]^T   (fp32 accumulate)
// block tile 128(M) x 256(N) x 32(K); 8 waves as 2(M) x 4(N); wave tile 64x64
// = 4x4 fragments -> 16 v_wmma per K-step. Grid: M fastest so blocks sharing
// the same W panel run consecutively (W panel stays L2-resident).
// ---------------------------------------------------------------------------
#define BM 128
#define BN 256
#define BK 32
#define BKP 40   // padded LDS row length (halves) to spread banks

__global__ __launch_bounds__(256) void gemm_kernel(const _Float16* __restrict__ A,
                                                   const _Float16* __restrict__ B,
                                                   float* __restrict__ C) {
    __shared__ _Float16 As[2][BM][BKP];   // 2 * 128*40*2B = 20.0 KB
    __shared__ _Float16 Bs[2][BN][BKP];   // 2 * 256*40*2B = 40.0 KB

    const int tid  = threadIdx.x;
    const int lane = tid & 31;
    const int wave = tid >> 5;      // 0..7
    const int wM   = wave >> 2;     // 0..1 -> M offset wM*64
    const int wN   = wave & 3;      // 0..3 -> N offset wN*64
    const int bm   = blockIdx.x * BM;   // M fastest-varying
    const int bn   = blockIdx.y * BN;

    // cooperative tile loads:
    //   A: 128 rows x 64B -> 2 threads/row, 32B each
    //   B: 256 rows x 64B -> 1 thread/row, 64B each
    const int lrowA = tid >> 1;          // 0..127
    const int lsegA = (tid & 1) * 16;    // 0 or 16 (halves)
    const _Float16* gA = A + (size_t)(bm + lrowA) * IN_DIM + lsegA;
    const _Float16* gB = B + (size_t)(bn + tid) * IN_DIM;

    v8f acc[4][4] = {};

    // prologue: tile 0 -> buffer 0
    {
        const uint4 a0 = *(const uint4*)gA;
        const uint4 a1 = *(const uint4*)(gA + 8);
        const uint4 b0 = *(const uint4*)gB;
        const uint4 b1 = *(const uint4*)(gB + 8);
        const uint4 b2 = *(const uint4*)(gB + 16);
        const uint4 b3 = *(const uint4*)(gB + 24);
        *(uint4*)&As[0][lrowA][lsegA]     = a0;
        *(uint4*)&As[0][lrowA][lsegA + 8] = a1;
        *(uint4*)&Bs[0][tid][0]  = b0;
        *(uint4*)&Bs[0][tid][8]  = b1;
        *(uint4*)&Bs[0][tid][16] = b2;
        *(uint4*)&Bs[0][tid][24] = b3;
    }
    __syncthreads();

    const int nsteps = IN_DIM / BK;   // 128
    uint4 ra0, ra1, rb0, rb1, rb2, rb3;

    for (int kt = 0; kt < nsteps; ++kt) {
        const int cur = kt & 1;
        const bool more = (kt + 1 < nsteps);

        if (more) {
            const _Float16* pa = gA + (size_t)(kt + 1) * BK;
            const _Float16* pb = gB + (size_t)(kt + 1) * BK;
            ra0 = *(const uint4*)pa;
            ra1 = *(const uint4*)(pa + 8);
            rb0 = *(const uint4*)pb;
            rb1 = *(const uint4*)(pb + 8);
            rb2 = *(const uint4*)(pb + 16);
            rb3 = *(const uint4*)(pb + 24);
            if (kt + 3 < nsteps) {
                __builtin_prefetch(gA + (size_t)(kt + 3) * BK, 0, 1);
                __builtin_prefetch(gB + (size_t)(kt + 3) * BK, 0, 1);
            }
        }

        // ---- fragment loads from LDS (ISA 16-bit A 16x32 / B 32x16 layouts) ----
        v16h afrag[4];
#pragma unroll
        for (int i = 0; i < 4; ++i) {
            const int row = wM * 64 + i * 16 + (lane & 15);
            const _Float16* ap = &As[cur][row][(lane >> 4) * 8];
            uint4* dst = (uint4*)&afrag[i];
            dst[0] = *(const uint4*)ap;          // K: base .. base+7
            dst[1] = *(const uint4*)(ap + 16);   // K: base+16 .. base+23
        }
        v16h bfrag[4];
#pragma unroll
        for (int j = 0; j < 4; ++j) {
            const int col = wN * 64 + j * 16 + (lane & 15);
            const _Float16* bp = &Bs[cur][col][(lane >> 4) * 8];
            uint4* dst = (uint4*)&bfrag[j];
            dst[0] = *(const uint4*)bp;          // K: base .. base+7
            dst[1] = *(const uint4*)(bp + 16);   // K: base+16 .. base+23
        }

#pragma unroll
        for (int i = 0; i < 4; ++i)
#pragma unroll
            for (int j = 0; j < 4; ++j)
                acc[i][j] = __builtin_amdgcn_wmma_f32_16x16x32_f16(
                    false, afrag[i], false, bfrag[j],
                    (short)0, acc[i][j], false, false);

        if (more) {
            const int nxt = cur ^ 1;
            *(uint4*)&As[nxt][lrowA][lsegA]     = ra0;
            *(uint4*)&As[nxt][lrowA][lsegA + 8] = ra1;
            *(uint4*)&Bs[nxt][tid][0]  = rb0;
            *(uint4*)&Bs[nxt][tid][8]  = rb1;
            *(uint4*)&Bs[nxt][tid][16] = rb2;
            *(uint4*)&Bs[nxt][tid][24] = rb3;
        }
        __syncthreads();
    }

    // ---- epilogue: f32 C/D layout; non-temporal stores keep W/x in L2 ----
    const int mbase = bm + wM * 64;
    const int nbase = bn + wN * 64;
#pragma unroll
    for (int i = 0; i < 4; ++i) {
#pragma unroll
        for (int j = 0; j < 4; ++j) {
            const int n     = nbase + j * 16 + (lane & 15);
            const int mrow0 = mbase + i * 16 + (lane >> 4) * 8;
#pragma unroll
            for (int v = 0; v < 8; ++v)
                __builtin_nontemporal_store(acc[i][j][v],
                    &C[(size_t)(mrow0 + v) * OUT_DIM + n]);
        }
    }
}

// ---------------------------------------------------------------------------
extern "C" void kernel_launch(void* const* d_in, const int* in_sizes, int n_in,
                              void* d_out, int out_size, void* d_ws, size_t ws_size,
                              hipStream_t stream) {
    const float* x          = (const float*)d_in[0];   // [2,1024,4096] fp32
    const float* scales     = (const float*)d_in[1];   // [11008] fp32
    const float* ortho_vals = (const float*)d_in[2];   // [NNZ] fp32
    const int*   base_pack  = (const int*)  d_in[3];   // [11008, 2048] int32
    const int*   ortho_rows = (const int*)  d_in[4];   // [NNZ] int32 (sorted)
    const int*   ortho_cols = (const int*)  d_in[5];   // [NNZ] int32
    float* out = (float*)d_out;                        // [2048, 11008] fp32
    const int nnz = in_sizes[2];

    // workspace layout: W f16 [OUT,IN] then xh f16 [TOKENS,IN]
    _Float16* W  = (_Float16*)d_ws;
    _Float16* xh = (_Float16*)((char*)d_ws + (size_t)OUT_DIM * IN_DIM * sizeof(_Float16));

    // 1) x -> f16
    {
        const int n4 = TOKENS * IN_DIM / 4;            // 2097152
        xcast_kernel<<<n4 / 256, 256, 0, stream>>>(x, xh);
    }
    // 2) dequant base weights -> W
    {
        const int nthr = OUT_DIM * HALF_IN / 4;        // 5636096
        dequant_kernel<<<nthr / 256, 256, 0, stream>>>(base_pack, scales, W);
    }
    // 3) deterministic sparse add
    scatter_kernel<<<(nnz + 255) / 256, 256, 0, stream>>>(ortho_vals, ortho_rows,
                                                          ortho_cols, W, nnz);
    // 4) WMMA GEMM: out = xh @ W^T   (grid: M fastest for W-panel L2 reuse)
    {
        dim3 grid(TOKENS / BM, OUT_DIM / BN);          // (16, 43)
        gemm_kernel<<<grid, 256, 0, stream>>>(xh, W, out);
    }
}
